// EmbedRefine_vishm2_71932112274123
// MI455X (gfx1250) — matrix operand order repaired and checked by reading, not compiled, not run on hardware
//
#include <hip/hip_runtime.h>
#include <hip/hip_bf16.h>
#include <math.h>

#define B_    8
#define D_    128
#define H_    152
#define W_    272
#define HW_   (H_*W_)
#define NDET_ 128
#define NTOK_ (B_*NDET_)   /* 1024 tokens */
#define HD_   16
#define DFF_  512

typedef __attribute__((ext_vector_type(16))) __bf16 v16bf;
typedef __attribute__((ext_vector_type(8)))  __bf16 v8bf;
typedef __attribute__((ext_vector_type(8)))  float  v8f;

// ---------------- WMMA operand loaders (match CDNA5 16-bit VGPR layouts) ---
// A (16x32 bf16): lane L row M=L%16; K elems = {half*8..+7, 16+half*8..+7}
__device__ __forceinline__ v16bf load_a16(const __bf16* src, int ld, int row0,
                                          int k0, int lane) {
  int m = lane & 15, half = lane >> 4;
  const __bf16* p = src + (size_t)(row0 + m) * ld + k0 + half * 8;
  v8bf lo = *(const v8bf*)p;
  v8bf hi = *(const v8bf*)(p + 16);
  v16bf r;
#pragma unroll
  for (int i = 0; i < 8; ++i) { r[i] = lo[i]; r[i + 8] = hi[i]; }
  return r;
}
// B (32x16 bf16) = W^T tile: lane L col N=L%16; K elems = {half*8..+7, 16+half*8..+7}
// (W row-major [n][k], so a B tile is two 16-byte chunks of row n)
__device__ __forceinline__ v16bf load_b16(const __bf16* w, int ld, int n0,
                                          int k0, int lane) {
  int nn = lane & 15, half = lane >> 4;
  const __bf16* p = w + (size_t)(n0 + nn) * ld + k0 + half * 8;
  v8bf lo = *(const v8bf*)p;
  v8bf hi = *(const v8bf*)(p + 16);
  v16bf r;
#pragma unroll
  for (int i = 0; i < 8; ++i) { r[i] = lo[i]; r[i + 8] = hi[i]; }
  return r;
}
// X(row-major, lda) @ W^T(W row-major, ldw), K multiple of 32, f32 accum
__device__ __forceinline__ v8f wmma_gemm(const __bf16* A, int lda, int row0,
                                         const __bf16* Wm, int ldw, int n0,
                                         int K, int lane, v8f c) {
  for (int k0 = 0; k0 < K; k0 += 32) {
    v16bf a = load_a16(A, lda, row0, k0, lane);
    v16bf b = load_b16(Wm, ldw, n0, k0, lane);
    c = __builtin_amdgcn_wmma_f32_16x16x32_bf16(false, a, false, b,
                                                (short)0, c, false, false);
  }
  return c;
}

// ---------------- Kernel 1: bulk copy x -> out (dominant HBM traffic) ------
__global__ void copy_x_kernel(const float4* __restrict__ src,
                              float4* __restrict__ dst, long n) {
  long stride = (long)gridDim.x * blockDim.x;
  for (long i = (long)blockIdx.x * blockDim.x + threadIdx.x; i < n; i += stride) {
    __builtin_prefetch(src + i + stride, 0, 1);   // global_prefetch_b8
    dst[i] = src[i];
  }
}

// ---------------- Kernel 2: fp32 -> bf16 weight conversion -----------------
__global__ void cvt_bf16_kernel(const float* __restrict__ s,
                                __bf16* __restrict__ d, int n) {
  int stride = gridDim.x * blockDim.x;
  for (int i = blockIdx.x * blockDim.x + threadIdx.x; i < n; i += stride)
    d[i] = (__bf16)s[i];
}

// ---------------- Kernel 3: adaptive 3x3 pool + q gather (L2-resident x) ---
__global__ __launch_bounds__(256) void pool_gather_kernel(
    const float* __restrict__ x, const float* __restrict__ vis,
    const int* __restrict__ dets, const int* __restrict__ inds,
    float* __restrict__ q_f, __bf16* __restrict__ q_bf,
    __bf16* __restrict__ k_bf) {
  int g = blockIdx.x;                 // token index 0..1023
  int b = g >> 7;
  int tid = threadIdx.x, lane = tid & 31, wid = tid >> 5;
  const int* dt = dets + (size_t)g * 4;
  int sx = dt[0], sy = dt[1], ex = dt[2], ey = dt[3];
  int Lx = ex - sx, Ly = ey - sy;
  int xs[3], xe[3], ys[3], ye[3];
#pragma unroll
  for (int i = 0; i < 3; ++i) {
    ys[i] = sy + (i * Ly) / 3;  ye[i] = sy + ((i + 1) * Ly + 2) / 3;
    xs[i] = sx + (i * Lx) / 3;  xe[i] = sx + ((i + 1) * Lx + 2) / 3;
  }
  // each wave: 16 channels; lanes sweep box columns (coalesced, L2 hits)
  for (int c9 = 0; c9 < 9; ++c9) {
    int cy = c9 / 3, cx = c9 % 3;
    int y0 = ys[cy], y1 = ye[cy], x0 = xs[cx], x1 = xe[cx];
    float inv = 1.0f / (float)((y1 - y0) * (x1 - x0));
    for (int dd = 0; dd < 16; ++dd) {
      int d = wid * 16 + dd;
      const float* base = x + ((size_t)(b * D_ + d)) * HW_;
      float s = 0.f;
      for (int h = y0; h < y1; ++h) {
        const float* row = base + h * W_;
        for (int w = x0 + lane; w < x1; w += 32) s += row[w];
      }
#pragma unroll
      for (int off = 16; off > 0; off >>= 1) s += __shfl_xor(s, off, 32);
      if (lane == 0)
        k_bf[((size_t)g * 9 + c9) * D_ + d] = (__bf16)(s * inv);
    }
  }
  // gathered query + sinusoidal visibility positional embedding
  if (tid < D_) {
    int d = tid;
    int p = inds[g];
    float xv = x[((size_t)(b * D_ + d)) * HW_ + p];
    float vv = vis[(size_t)b * HW_ + p];
    int vidx = (int)(vv * 10.0f);
    float freq = powf(10000.f, (float)(2 * (d >> 1)) / (float)D_);
    float arg = (float)vidx * 0.1f / freq;
    float pe = (d & 1) ? cosf(arg) : sinf(arg);
    float qv = xv + pe;
    q_f[(size_t)g * D_ + d] = qv;
    q_bf[(size_t)g * D_ + d] = (__bf16)qv;
  }
}

// ---------------- Kernel 4: fused attention + FFN per 16-token tile --------
__global__ __launch_bounds__(256) void token_xform_kernel(
    const __bf16* __restrict__ q_bf, const __bf16* __restrict__ k_bf,
    const float* __restrict__ q_f,
    const __bf16* __restrict__ wqkv, const __bf16* __restrict__ wo,
    const __bf16* __restrict__ w1, const __bf16* __restrict__ w2,
    const float* __restrict__ bqkv, const float* __restrict__ bo,
    const float* __restrict__ b1, const float* __restrict__ b2,
    const float* __restrict__ g2, const float* __restrict__ be2,
    const float* __restrict__ g3, const float* __restrict__ be3,
    const int* __restrict__ inds, float* __restrict__ out) {
  extern __shared__ __align__(16) char smem[];
  float*  kp  = (float*)smem;            // 144 x 128 f32
  float*  vp  = kp + 144 * D_;           // 144 x 128 f32
  float*  qp  = vp + 144 * D_;           // 16 x 128 f32
  float*  tb  = qp + 16 * D_;            // 16 x 128 f32
  __bf16* aob = (__bf16*)(tb + 16 * D_); // 16 x 128 bf16
  __bf16* tbf = aob + 16 * D_;           // 16 x 128 bf16
  __bf16* h1b = tbf + 16 * D_;           // 16 x 512 bf16

  int tid = threadIdx.x, lane = tid & 31, wid = tid >> 5;
  int t0 = blockIdx.x * 16;
  int nloc = lane & 15, half = lane >> 4;

  // ---- Q projection (M=16, N=128, K=128)
  {
    int n0 = wid * 16;
    v8f c = {};
    c = wmma_gemm(q_bf, D_, t0, wqkv, D_, n0, D_, lane, c);
    float bias = bqkv[n0 + nloc];
#pragma unroll
    for (int v = 0; v < 8; ++v)
      qp[(v + 8 * half) * D_ + n0 + nloc] = c[v] + bias;
  }
  // ---- K / V projections (M=144 in 9 tiles)
  const __bf16* Wk = wqkv + D_ * D_;
  const __bf16* Wv = wqkv + 2 * D_ * D_;
  for (int mt = 0; mt < 9; ++mt) {
    int n0 = wid * 16;
    int row0 = t0 * 9 + mt * 16;
    v8f ck = {}; ck = wmma_gemm(k_bf, D_, row0, Wk, D_, n0, D_, lane, ck);
    v8f cv = {}; cv = wmma_gemm(k_bf, D_, row0, Wv, D_, n0, D_, lane, cv);
    float bk = bqkv[D_ + n0 + nloc];
    float bv = bqkv[2 * D_ + n0 + nloc];
#pragma unroll
    for (int v = 0; v < 8; ++v) {
      int m = mt * 16 + v + 8 * half;
      kp[m * D_ + n0 + nloc] = ck[v] + bk;
      vp[m * D_ + n0 + nloc] = cv[v] + bv;
    }
  }
  __syncthreads();
  // ---- attention: 1 query x 9 keys per (token, head); VALU
  if (tid < 128) {
    int lt = tid >> 3, hh = tid & 7;
    const float* qv = qp + lt * D_ + hh * HD_;
    float att[9]; float mx = -1e30f;
    for (int s = 0; s < 9; ++s) {
      const float* kr = kp + (lt * 9 + s) * D_ + hh * HD_;
      float dt = 0.f;
#pragma unroll
      for (int e = 0; e < HD_; ++e) dt += qv[e] * kr[e];
      att[s] = dt * 0.25f;                      // 1/sqrt(hd=16)
      mx = fmaxf(mx, att[s]);
    }
    float den = 0.f;
    for (int s = 0; s < 9; ++s) { att[s] = __expf(att[s] - mx); den += att[s]; }
    float r = 1.f / den;
#pragma unroll
    for (int e = 0; e < HD_; ++e) {
      float acc = 0.f;
      for (int s = 0; s < 9; ++s)
        acc += att[s] * vp[(lt * 9 + s) * D_ + hh * HD_ + e];
      aob[lt * D_ + hh * HD_ + e] = (__bf16)(acc * r);
    }
  }
  __syncthreads();
  // ---- output projection + residual with gathered q (fp32)
  {
    int n0 = wid * 16;
    v8f c = {};
    c = wmma_gemm(aob, D_, 0, wo, D_, n0, D_, lane, c);
    float bias = bo[n0 + nloc];
#pragma unroll
    for (int v = 0; v < 8; ++v) {
      int m = v + 8 * half;
      tb[m * D_ + n0 + nloc] =
          q_f[(size_t)(t0 + m) * D_ + n0 + nloc] + c[v] + bias;
    }
  }
  __syncthreads();
  // ---- LayerNorm 2 (16 lanes per token, xor-shuffle reduction) + bf16 copy
  {
    int lt = tid >> 4, c0 = tid & 15;
    float* row = tb + lt * D_;
    float v[8], s = 0.f, s2 = 0.f;
#pragma unroll
    for (int j = 0; j < 8; ++j) { v[j] = row[c0 + 16 * j]; s += v[j]; s2 += v[j] * v[j]; }
#pragma unroll
    for (int off = 1; off < 16; off <<= 1) {
      s += __shfl_xor(s, off, 32); s2 += __shfl_xor(s2, off, 32);
    }
    float mean = s * (1.f / 128.f);
    float rstd = rsqrtf(s2 * (1.f / 128.f) - mean * mean + 1e-5f);
#pragma unroll
    for (int j = 0; j < 8; ++j) {
      int cc = c0 + 16 * j;
      float t = (v[j] - mean) * rstd * g2[cc] + be2[cc];
      row[cc] = t;
      tbf[lt * D_ + cc] = (__bf16)t;
    }
  }
  __syncthreads();
  // ---- FFN1 (N=512: 4 N-tiles per wave) + ReLU -> bf16
#pragma unroll
  for (int j = 0; j < 4; ++j) {
    int n0 = (wid + 8 * j) * 16;
    v8f c = {};
    c = wmma_gemm(tbf, D_, 0, w1, D_, n0, D_, lane, c);
    float bias = b1[n0 + nloc];
#pragma unroll
    for (int v = 0; v < 8; ++v) {
      int m = v + 8 * half;
      h1b[m * DFF_ + n0 + nloc] = (__bf16)fmaxf(c[v] + bias, 0.f);
    }
  }
  __syncthreads();
  // ---- FFN2 (K=512) + residual in place
  {
    int n0 = wid * 16;
    v8f c = {};
    c = wmma_gemm(h1b, DFF_, 0, w2, DFF_, n0, DFF_, lane, c);
    float bias = b2[n0 + nloc];
#pragma unroll
    for (int v = 0; v < 8; ++v) {
      int m = v + 8 * half;
      tb[m * D_ + n0 + nloc] += c[v] + bias;
    }
  }
  __syncthreads();
  // ---- LayerNorm 3 (in place)
  {
    int lt = tid >> 4, c0 = tid & 15;
    float* row = tb + lt * D_;
    float v[8], s = 0.f, s2 = 0.f;
#pragma unroll
    for (int j = 0; j < 8; ++j) { v[j] = row[c0 + 16 * j]; s += v[j]; s2 += v[j] * v[j]; }
#pragma unroll
    for (int off = 1; off < 16; off <<= 1) {
      s += __shfl_xor(s, off, 32); s2 += __shfl_xor(s2, off, 32);
    }
    float mean = s * (1.f / 128.f);
    float rstd = rsqrtf(s2 * (1.f / 128.f) - mean * mean + 1e-5f);
#pragma unroll
    for (int j = 0; j < 8; ++j) {
      int cc = c0 + 16 * j;
      row[cc] = (v[j] - mean) * rstd * g3[cc] + be3[cc];
    }
  }
  __syncthreads();
  // ---- scatter tokens back into the (B,D,H,W) output
  for (int i = tid; i < 16 * D_; i += 256) {
    int lt = i >> 7, d = i & 127;
    int g = t0 + lt;
    int b = g >> 7;
    int p = inds[g];
    out[((size_t)(b * D_ + d)) * HW_ + p] = tb[lt * D_ + d];
  }
}

// ---------------------------------------------------------------------------
extern "C" void kernel_launch(void* const* d_in, const int* in_sizes, int n_in,
                              void* d_out, int out_size, void* d_ws,
                              size_t ws_size, hipStream_t stream) {
  (void)in_sizes; (void)n_in; (void)out_size; (void)ws_size;
  const float* x    = (const float*)d_in[0];
  const float* vis  = (const float*)d_in[1];
  const int*   dets = (const int*)d_in[2];
  const int*   inds = (const int*)d_in[3];
  const float* Wqkv = (const float*)d_in[4];
  const float* bqkv = (const float*)d_in[5];
  const float* Wo   = (const float*)d_in[6];
  const float* bo   = (const float*)d_in[7];
  const float* W1   = (const float*)d_in[8];
  const float* b1   = (const float*)d_in[9];
  const float* W2   = (const float*)d_in[10];
  const float* b2   = (const float*)d_in[11];
  const float* g2   = (const float*)d_in[12];
  const float* be2  = (const float*)d_in[13];
  const float* g3   = (const float*)d_in[14];
  const float* be3  = (const float*)d_in[15];
  float* out = (float*)d_out;

  // workspace layout (bytes)
  char* ws = (char*)d_ws;
  float*  q_f     = (float*)(ws + 0);          // 1024*128 f32  = 524288
  __bf16* q_bf    = (__bf16*)(ws + 524288);    // 1024*128 bf16 = 262144
  __bf16* k_bf    = (__bf16*)(ws + 786432);    // 1024*9*128    = 2359296
  __bf16* wqkv_bf = (__bf16*)(ws + 3145728);   // 384*128       =   98304
  __bf16* wo_bf   = (__bf16*)(ws + 3244032);   // 128*128       =   32768
  __bf16* w1_bf   = (__bf16*)(ws + 3276800);   // 512*128       =  131072
  __bf16* w2_bf   = (__bf16*)(ws + 3407872);   // 128*512       =  131072

  cvt_bf16_kernel<<<192, 256, 0, stream>>>(Wqkv, wqkv_bf, 3 * D_ * D_);
  cvt_bf16_kernel<<<64, 256, 0, stream>>>(Wo, wo_bf, D_ * D_);
  cvt_bf16_kernel<<<256, 256, 0, stream>>>(W1, w1_bf, DFF_ * D_);
  cvt_bf16_kernel<<<256, 256, 0, stream>>>(W2, w2_bf, D_ * DFF_);

  long n4 = (long)B_ * D_ * HW_ / 4;
  copy_x_kernel<<<4096, 256, 0, stream>>>((const float4*)x, (float4*)out, n4);

  pool_gather_kernel<<<NTOK_, 256, 0, stream>>>(x, vis, dets, inds, q_f, q_bf,
                                                k_bf);

  const int SMEM = (2 * 144 * D_ + 2 * 16 * D_) * (int)sizeof(float) +
                   (2 * 16 * D_ + 16 * DFF_) * (int)sizeof(__bf16); // 188416 B
  hipFuncSetAttribute(reinterpret_cast<const void*>(token_xform_kernel),
                      hipFuncAttributeMaxDynamicSharedMemorySize, SMEM);
  token_xform_kernel<<<NTOK_ / 16, 256, SMEM, stream>>>(
      q_bf, k_bf, q_f, wqkv_bf, wo_bf, w1_bf, w2_bf, bqkv, bo, b1, b2, g2, be2,
      g3, be3, inds, out);
}
